// LearnableDemosaick_90417651516506
// MI455X (gfx1250) — matrix operand order repaired and checked by reading, not compiled, not run on hardware
//
#include <hip/hip_runtime.h>
#include <hip/hip_bf16.h>

typedef __attribute__((ext_vector_type(16))) _Float16 v16h;
typedef __attribute__((ext_vector_type(8)))  float    v8f;

#define FSIZE   5
#define PADW    2
#define KF      8
#define BATCH   16
#define IMG_H   512
#define IMG_W   512
#define TW      128             // output tile width per block
#define TH      8               // output tile height per block (one row per wave)
#define IN_W    (TW + 2*PADW)   // 132
#define IN_H    (TH + 2*PADW)   // 12
#define ZBASE   (IN_H * IN_W)   // 1584: start of zeroed LDS region
#define ZSIZE   128             // covers base+offs = ZBASE + t*16, t in [0,7]
#define NTHREADS 256

// xor-16 half swap in wave32 via v_permlanex16 (identity lane selects)
__device__ __forceinline__ float permswap16(float x) {
    int xi = __builtin_bit_cast(int, x);
    int oi = __builtin_amdgcn_permlanex16(xi, xi, 0x76543210, (int)0xfedcba98,
                                          false, false);
    return __builtin_bit_cast(float, oi);
}

// D = A(filters 16x32 f16) x B(patches 32x16 f16) + 0
//   rows 0..7  of D = sel scores,  rows 8..15 = green candidates
//   col n of D = pixel n of the 16-pixel tile
__global__ __launch_bounds__(NTHREADS)
void demosaick_wmma_kernel(const float* __restrict__ mosaick,
                           const float* __restrict__ sel_filts,
                           const float* __restrict__ green_filts,
                           float* __restrict__ out)
{
    __shared__ _Float16 smem[IN_H * IN_W + ZSIZE];

    const int tid  = threadIdx.x;
    const int lane = tid & 31;
    const int wave = tid >> 5;

    const int x0 = blockIdx.x * TW;
    const int y0 = blockIdx.y * TH;
    const float* img    = mosaick + (size_t)blockIdx.z * IMG_H * IMG_W;
    float*       outimg = out     + (size_t)blockIdx.z * IMG_H * IMG_W;

    // ---- Stage edge-clamped (TH+4)x(TW+4) window into LDS as f16 ----
    for (int i = tid; i < IN_H * IN_W; i += NTHREADS) {
        int iy = i / IN_W;
        int ix = i - iy * IN_W;
        int gy = y0 + iy - PADW; gy = gy < 0 ? 0 : (gy >= IMG_H ? IMG_H - 1 : gy);
        int gx = x0 + ix - PADW; gx = gx < 0 ? 0 : (gx >= IMG_W ? IMG_W - 1 : gx);
        smem[i] = (_Float16)img[gy * IMG_W + gx];
    }
    if (tid < ZSIZE) smem[ZBASE + tid] = (_Float16)0.0f;   // zero pad region

    const int r  = wave;           // this wave's output row inside the tile
    const int yg = y0 + r;
    const int nx = lane & 15;      // pixel column within 16-pixel WMMA tile

    // ---- A fragment: filters, 16x32 f16 ----
    // lane L holds row M = L&15; slot j -> K = j + (L>=16?16:0)
    const int M    = lane & 15;
    const float* fsrc = (M < 8) ? sel_filts : green_filts;
    const int fcol = M & 7;
    const int kofA = (lane & 16) ? 16 : 0;
    v16h afrag;
    #pragma unroll
    for (int j = 0; j < 16; ++j) {
        int k = j + kofA;
        int e = (k < 25) ? k : 0;
        float fv = fsrc[e * KF + fcol];
        afrag[j] = (k < 25) ? (_Float16)fv : (_Float16)0.0f;
    }

    // ---- Per-lane LDS tap offsets for B fragment ----
    // lane L holds col N = L&15 (pixel); slot j -> K = j + (L>=16?16:0)
    // invalid slots point into the zero region (independent of tile index t)
    const int kofB = (lane & 16) ? 16 : 0;
    int offs[16];
    #pragma unroll
    for (int j = 0; j < 16; ++j) {
        int k = j + kofB;
        if (k < 25) {
            int fy = k / FSIZE;
            int fx = k - fy * FSIZE;
            offs[j] = fy * IN_W + fx;
        } else {
            offs[j] = ZBASE - (r * IN_W + nx);   // base+offs = ZBASE + t*16
        }
    }

    __syncthreads();

    #pragma unroll
    for (int t = 0; t < TW / 16; ++t) {
        const int base = r * IN_W + t * 16 + nx;

        v16h bfrag;
        #pragma unroll
        for (int j = 0; j < 16; ++j)
            bfrag[j] = smem[base + offs[j]];     // unconditional: pads read 0

        v8f c = {};
        v8f d = __builtin_amdgcn_wmma_f32_16x16x32_f16(
                    false, afrag, false, bfrag, (short)0, c, false, false);

        // lanes 0..15: d[v] = sel_v(pixel nx); partner lane 16..31 holds grn_v.
        // Only lanes 0..15 produce a meaningful result (upper half unused).
        float grn[8];
        #pragma unroll
        for (int v = 0; v < 8; ++v) grn[v] = permswap16(d[v]);

        float mx = d[0];
        #pragma unroll
        for (int v = 1; v < 8; ++v) mx = fmaxf(mx, d[v]);
        float num = 0.0f, den = 0.0f;
        #pragma unroll
        for (int v = 0; v < 8; ++v) {
            float e = __expf(d[v] - mx);
            den += e;
            num += e * grn[v];
        }
        float interp = num / den;

        if (lane < 16) {
            int xg  = x0 + t * 16 + lane;
            int idx = yg * IMG_W + xg;
            float mos = img[idx];                          // exact f32 copy path
            outimg[idx] = (((yg ^ xg) & 1) == 0) ? mos : interp;
        }
    }
}

extern "C" void kernel_launch(void* const* d_in, const int* in_sizes, int n_in,
                              void* d_out, int out_size, void* d_ws, size_t ws_size,
                              hipStream_t stream) {
    const float* mosaick     = (const float*)d_in[0];
    const float* sel_filts   = (const float*)d_in[1];
    const float* green_filts = (const float*)d_in[2];
    float* out = (float*)d_out;

    dim3 grid(IMG_W / TW, IMG_H / TH, BATCH);   // 4 x 64 x 16
    dim3 block(NTHREADS);
    hipLaunchKernelGGL(demosaick_wmma_kernel, grid, block, 0, stream,
                       mosaick, sel_filts, green_filts, out);
}
